// MP_TFWA_6210522710308
// MI455X (gfx1250) — matrix operand order repaired and checked by reading, not compiled, not run on hardware
//
#include <hip/hip_runtime.h>
#include <hip/hip_bf16.h>
#include <math.h>

typedef _Float16 half_t;
typedef __attribute__((ext_vector_type(16))) _Float16 v16h;
typedef __attribute__((ext_vector_type(8)))  _Float16 h8;
typedef __attribute__((ext_vector_type(8)))  float    v8f;

#define LDA 40   // LDS row stride in halfs (32 data + 8 pad -> 80B, bank-spread, 16B aligned)
#define SHUF16(lo, hi) __builtin_shufflevector(lo, hi, 0,1,2,3,4,5,6,7,8,9,10,11,12,13,14,15)

// ---------------------------------------------------------------------------
// Strided batched GEMM, f16 inputs, f32 accum via V_WMMA_F32_16X16X32_F16.
//   C[b,m,n] = alpha * sum_k A[b,m,k]*B[b,k,n] + bias[n]
// A elem: A + b*abat + m*a_sm + k*a_sk ;  B elem: B + b*bbat + k*b_sk + n*b_sn
// Block: 256 threads = 8 waves, 128(M) x 64(N) tile, K stepped by 32.
// Wave (wm 0..3, wn 0..1): 32x32 output = 4 accumulators (2 A-frag x 2 B-frag).
// Double-buffered LDS; global fetch of tile s+1 overlaps compute of tile s.
// ---------------------------------------------------------------------------
__global__ __launch_bounds__(256)
void wmma_gemm_k(const half_t* __restrict__ A, long a_sm, long a_sk, long abat,
                 const half_t* __restrict__ Bm, long b_sk, long b_sn, long bbat,
                 const float* __restrict__ bias,
                 half_t* __restrict__ outH, float* __restrict__ outF,
                 long o_sm, long obat,
                 int M, int N, int K, float alpha)
{
    __shared__ __align__(16) _Float16 Al[2][128 * LDA];  // [buf][m_local][k_local]
    __shared__ __align__(16) _Float16 Bl[2][64 * LDA];   // [buf][n_local][k_local]

    const int tid   = threadIdx.x;
    const int batch = blockIdx.z;
    const half_t* Ab = A  + (long)batch * abat;
    const half_t* Bb = Bm + (long)batch * bbat;

    const int tile_m0 = blockIdx.y * 128;
    const int tile_n0 = blockIdx.x * 64;

    // Uniform fast-path eligibility (16B-aligned vector copies, full tiles).
    const bool a_kf = (a_sk == 1) && ((a_sm & 7) == 0) && (tile_m0 + 128 <= M);
    const bool a_mf = (a_sm == 1) && ((a_sk & 7) == 0) && (tile_m0 + 128 <= M);
    const bool b_nf = (b_sn == 1) && ((b_sk & 7) == 0) && (tile_n0 + 64 <= N);
    const bool b_kf = (b_sk == 1) && ((b_sn & 7) == 0) && (tile_n0 + 64 <= N);

    h8 aR0, aR1, bR;   // register staging for the in-flight tile

    auto fetch = [&](int k0) {
        const bool kfull = (k0 + 32 <= K);
        // ---- A tile 128x32 (2 x h8 per thread) ----
        if (a_kf && kfull) {
            const int ml = tid >> 1, kl = (tid & 1) * 16;
            const half_t* s = Ab + (long)(tile_m0 + ml) * a_sm + (k0 + kl);
            __builtin_prefetch(s + 32, 0, 1);
            aR0 = *(const h8*)s;
            aR1 = *(const h8*)(s + 8);
        } else if (a_mf && kfull) {
            const int kl = tid >> 3, mb = (tid & 7) * 16;
            const half_t* s = Ab + (long)(k0 + kl) * a_sk + (tile_m0 + mb);
            aR0 = *(const h8*)s;
            aR1 = *(const h8*)(s + 8);
        } else {
            #pragma unroll
            for (int it = 0; it < 16; ++it) {
                const int idx = it * 256 + tid;         // 128*32 elements
                const int ml = idx >> 5, kl = idx & 31;
                const int m = tile_m0 + ml, k = k0 + kl;
                const int mc = m < M ? m : (M - 1);
                const int kc = k < K ? k : (K - 1);
                half_t v = Ab[(long)mc * a_sm + (long)kc * a_sk];
                v = (k < K) ? v : (half_t)0;
                if (it < 8) aR0[it] = v; else aR1[it - 8] = v;
            }
        }
        // ---- B tile 64x32 (1 x h8 per thread) ----
        if (b_nf && kfull) {
            const int kl = tid >> 3, nb = (tid & 7) * 8;
            bR = *(const h8*)(Bb + (long)(k0 + kl) * b_sk + (tile_n0 + nb));
        } else if (b_kf && kfull) {
            const int nl = tid >> 2, kl = (tid & 3) * 8;
            bR = *(const h8*)(Bb + (long)(tile_n0 + nl) * b_sn + (k0 + kl));
        } else {
            #pragma unroll
            for (int it = 0; it < 8; ++it) {
                const int idx = it * 256 + tid;         // 64*32 elements
                const int nl = idx >> 5, kl = idx & 31;
                const int n = tile_n0 + nl, k = k0 + kl;
                const int nc = n < N ? n : (N - 1);
                const int kc = k < K ? k : (K - 1);
                half_t v = Bb[(long)kc * b_sk + (long)nc * b_sn];
                bR[it] = (k < K) ? v : (half_t)0;
            }
        }
    };

    auto commit = [&](int buf, int k0) {
        const bool kfull = (k0 + 32 <= K);
        _Float16* Ad = Al[buf];
        _Float16* Bd = Bl[buf];
        if (a_kf && kfull) {
            const int ml = tid >> 1, kl = (tid & 1) * 16;
            *(h8*)(&Ad[ml * LDA + kl])     = aR0;
            *(h8*)(&Ad[ml * LDA + kl + 8]) = aR1;
        } else if (a_mf && kfull) {
            const int kl = tid >> 3, mb = (tid & 7) * 16;
            #pragma unroll
            for (int e = 0; e < 8; ++e) {
                Ad[(mb + e) * LDA + kl]     = aR0[e];
                Ad[(mb + 8 + e) * LDA + kl] = aR1[e];
            }
        } else {
            #pragma unroll
            for (int it = 0; it < 16; ++it) {
                const int idx = it * 256 + tid;
                const int ml = idx >> 5, kl = idx & 31;
                Ad[ml * LDA + kl] = (it < 8) ? aR0[it] : aR1[it - 8];
            }
        }
        if (b_nf && kfull) {
            const int kl = tid >> 3, nb = (tid & 7) * 8;
            #pragma unroll
            for (int e = 0; e < 8; ++e) Bd[(nb + e) * LDA + kl] = bR[e];
        } else if (b_kf && kfull) {
            const int nl = tid >> 2, kl = (tid & 3) * 8;
            *(h8*)(&Bd[nl * LDA + kl]) = bR;
        } else {
            #pragma unroll
            for (int it = 0; it < 8; ++it) {
                const int idx = it * 256 + tid;
                const int nl = idx >> 5, kl = idx & 31;
                Bd[nl * LDA + kl] = bR[it];
            }
        }
    };

    const int lane    = tid & 31;
    const int wave    = tid >> 5;
    const int wm      = wave >> 1;    // 0..3 -> m base = wm*32
    const int wn      = wave & 1;     // 0..1 -> n base = wn*32
    const int half_id = lane >> 4;    // 0 or 1
    const int lrow    = lane & 15;

    v8f acc00 = {}, acc01 = {}, acc10 = {}, acc11 = {};
    const int steps = (K + 31) / 32;

    fetch(0);
    commit(0, 0);
    __syncthreads();

    for (int s = 0; s < steps; ++s) {
        const int cur  = s & 1;
        const int k0n  = (s + 1) * 32;
        const bool more = (s + 1) < steps;
        if (more) fetch(k0n);   // global loads in flight during compute

        const _Float16* Ad = Al[cur];
        const _Float16* Bd = Bl[cur];

        // B 32x16 frags: lane holds col n=lrow, K run = half_id*16 .. +15
        const _Float16* bp0 = &Bd[(wn * 32 + lrow) * LDA + half_id * 16];
        v16h bf0 = SHUF16(*(const h8*)bp0, *(const h8*)(bp0 + 8));
        const _Float16* bp1 = &Bd[(wn * 32 + 16 + lrow) * LDA + half_id * 16];
        v16h bf1 = SHUF16(*(const h8*)bp1, *(const h8*)(bp1 + 8));

        // A 16x32 frags: lane row, K run = half_id*16 .. +15
        const _Float16* ap0 = &Ad[(wm * 32 + lrow) * LDA + half_id * 16];
        v16h af0 = SHUF16(*(const h8*)ap0, *(const h8*)(ap0 + 8));
        const _Float16* ap1 = &Ad[(wm * 32 + 16 + lrow) * LDA + half_id * 16];
        v16h af1 = SHUF16(*(const h8*)ap1, *(const h8*)(ap1 + 8));

        acc00 = __builtin_amdgcn_wmma_f32_16x16x32_f16(false, af0, false, bf0, (short)0, acc00, false, false);
        acc01 = __builtin_amdgcn_wmma_f32_16x16x32_f16(false, af0, false, bf1, (short)0, acc01, false, false);
        acc10 = __builtin_amdgcn_wmma_f32_16x16x32_f16(false, af1, false, bf0, (short)0, acc10, false, false);
        acc11 = __builtin_amdgcn_wmma_f32_16x16x32_f16(false, af1, false, bf1, (short)0, acc11, false, false);

        if (more) commit(1 - cur, k0n);
        __syncthreads();
    }

    // ---- epilogue: C 16x16 f32, lane<16 -> M=r, lane>=16 -> M=r+8; N=lrow ----
    auto cstore = [&](const v8f& ac, int rbase, int col) {
        if (col < N) {
            const float bv = bias ? bias[col] : 0.0f;
            #pragma unroll
            for (int r = 0; r < 8; ++r) {
                const int orow = rbase + half_id * 8 + r;
                if (orow < M) {
                    const float v = ac[r] * alpha + bv;
                    if (outF) outF[(long)batch * obat + (long)orow * o_sm + col] = v;
                    else      outH[(long)batch * obat + (long)orow * o_sm + col] = (half_t)v;
                }
            }
        }
    };
    const int gm  = tile_m0 + wm * 32;
    const int gn0 = tile_n0 + wn * 32 + lrow;
    cstore(acc00, gm,      gn0);
    cstore(acc01, gm,      gn0 + 16);
    cstore(acc10, gm + 16, gn0);
    cstore(acc11, gm + 16, gn0 + 16);
}

// Row-wise softmax in-place on f16 data; one 256-thread block per row.
__global__ __launch_bounds__(256)
void softmax_rows_k(half_t* __restrict__ data, int L)
{
    __shared__ float red[256];
    half_t* p = data + (long)blockIdx.x * L;
    const int tid = threadIdx.x;

    float mx = -3.0e38f;
    for (int i = tid; i < L; i += 256) mx = fmaxf(mx, (float)p[i]);
    red[tid] = mx; __syncthreads();
    for (int s = 128; s > 0; s >>= 1) {
        if (tid < s) red[tid] = fmaxf(red[tid], red[tid + s]);
        __syncthreads();
    }
    mx = red[0]; __syncthreads();

    float sum = 0.0f;
    for (int i = tid; i < L; i += 256) {
        const float e = __expf((float)p[i] - mx);
        p[i] = (half_t)e;
        sum += e;
    }
    red[tid] = sum; __syncthreads();
    for (int s = 128; s > 0; s >>= 1) {
        if (tid < s) red[tid] += red[tid + s];
        __syncthreads();
    }
    const float inv = 1.0f / red[0];
    for (int i = tid; i < L; i += 256) p[i] = (half_t)((float)p[i] * inv);
}

__global__ void f32_to_f16_k(const float* __restrict__ in, half_t* __restrict__ out, long n)
{
    const long i = (long)blockIdx.x * blockDim.x + threadIdx.x;
    if (i < n) out[i] = (half_t)in[i];
}

// f32 (rows x cols) -> f16 with padded row stride (for 16B-aligned fast paths).
__global__ void pad_cvt_k(const float* __restrict__ in, half_t* __restrict__ out,
                          int cols, int ostride, long n)
{
    const long i = (long)blockIdx.x * blockDim.x + threadIdx.x;
    if (i >= n) return;
    const long r = i / cols;
    const int  c = (int)(i % cols);
    out[r * ostride + c] = (half_t)in[i];
}

// x = tokens[:, 1:, :]  ((B, D+1, H) f32 -> (B, D, H) f16), drops CLS row.
__global__ void gather_x_k(const float* __restrict__ tok, half_t* __restrict__ xh,
                           int D, int H, long n)
{
    const long i = (long)blockIdx.x * blockDim.x + threadIdx.x;
    if (i >= n) return;
    const int  h = (int)(i % H);
    const long t = i / H;
    const int  d = (int)(t % D);
    const long b = t / D;
    xh[i] = (half_t)tok[(b * (D + 1) + (d + 1)) * (long)H + h];
}

// Fills cols [0, 2304) of concat (B x 5304): mrc_cls | co_cls | mask_vec.
__global__ void build_concat_k(const float* __restrict__ mrc, const float* __restrict__ co,
                               const float* __restrict__ pl, const int* __restrict__ midx,
                               half_t* __restrict__ cat, long n)
{
    const long i = (long)blockIdx.x * blockDim.x + threadIdx.x;
    if (i >= n) return;
    const int col = (int)(i % 2304);
    const int b   = (int)(i / 2304);
    float v;
    if (col < 768)       v = mrc[(long)b * 289 * 768 + col];
    else if (col < 1536) v = co[(long)b * 257 * 768 + (col - 768)];
    else {
        const int r = midx[b * 2 + 1];
        v = pl[((long)b * 289 + r) * 768 + (col - 1536)];
    }
    cat[(long)b * 5304 + col] = (half_t)v;
}

// ---------------------------------------------------------------------------
static void gemm(hipStream_t s,
                 const half_t* A, long a_sm, long a_sk, long abat,
                 const half_t* B, long b_sk, long b_sn, long bbat,
                 const float* bias, half_t* outH, float* outF, long o_sm, long obat,
                 int M, int N, int K, int batch, float alpha)
{
    dim3 grid((N + 63) / 64, (M + 127) / 128, batch);
    wmma_gemm_k<<<grid, 256, 0, s>>>(A, a_sm, a_sk, abat, B, b_sk, b_sn, bbat,
                                     bias, outH, outF, o_sm, obat, M, N, K, alpha);
}

extern "C" void kernel_launch(void* const* d_in, const int* in_sizes, int n_in,
                              void* d_out, int out_size, void* d_ws, size_t ws_size,
                              hipStream_t stream)
{
    (void)in_sizes; (void)n_in; (void)out_size; (void)ws_size;
    const int H = 768, B = 64;
    const int Ds[3] = {288, 256, 288};
    const int Dmax = 288;
    const int PW = 128;   // padded column stride for the 100-wide tw/fw tensors

    size_t off = 0;
    auto alloc_h = [&](size_t elems) -> half_t* {
        half_t* p = (half_t*)((char*)d_ws + off);
        off = (off + elems * sizeof(half_t) + 255) & ~(size_t)255;
        return p;
    };
    auto cvt = [&](int idx, size_t n) -> half_t* {
        half_t* dst = alloc_h(n);
        f32_to_f16_k<<<(int)((n + 255) / 256), 256, 0, stream>>>((const float*)d_in[idx], dst, (long)n);
        return dst;
    };
    auto cvt_pad = [&](int idx, size_t rows, int cols, int ostride) -> half_t* {
        half_t* dst = alloc_h(rows * ostride);
        const long n = (long)rows * cols;
        pad_cvt_k<<<(int)((n + 255) / 256), 256, 0, stream>>>((const float*)d_in[idx], dst, cols, ostride, n);
        return dst;
    };

    // ---- weights -> f16 (d_in: 4 tensors, 18 leaves/branch, fc1 W,b, fc2 W,b)
    half_t *Wtq[3], *Wtk[3], *Wtv[3], *Wfq[3], *Wfk[3], *Wfv[3], *WTW[3], *WFW[3], *Wwov[3];
    const float *btq[3], *btk[3], *btv[3], *bfq[3], *bfk[3], *bfv[3], *bTW[3], *bFW[3], *bwov[3];
    for (int bi = 0; bi < 3; ++bi) {
        const int p = 4 + 18 * bi;
        const size_t D = (size_t)Ds[bi];
        Wtq[bi]  = cvt(p + 0,  (size_t)H * H);        btq[bi]  = (const float*)d_in[p + 1];
        Wtk[bi]  = cvt(p + 2,  (size_t)H * H);        btk[bi]  = (const float*)d_in[p + 3];
        Wtv[bi]  = cvt(p + 4,  (size_t)H * H);        btv[bi]  = (const float*)d_in[p + 5];
        Wfq[bi]  = cvt(p + 6,  D * D);                bfq[bi]  = (const float*)d_in[p + 7];
        Wfk[bi]  = cvt(p + 8,  D * D);                bfk[bi]  = (const float*)d_in[p + 9];
        Wfv[bi]  = cvt(p + 10, D * D);                bfv[bi]  = (const float*)d_in[p + 11];
        WTW[bi]  = cvt_pad(p + 12, (size_t)H, 100, PW); bTW[bi] = (const float*)d_in[p + 13];
        WFW[bi]  = cvt_pad(p + 14, (size_t)H, 100, PW); bFW[bi] = (const float*)d_in[p + 15];
        Wwov[bi] = cvt(p + 16, (size_t)10000 * 1000); bwov[bi] = (const float*)d_in[p + 17];
    }
    half_t* Wfc1 = cvt(58, (size_t)5304 * 768);  const float* bfc1 = (const float*)d_in[59];
    half_t* Wfc2 = cvt(60, (size_t)768 * 5);     const float* bfc2 = (const float*)d_in[61];

    // ---- activation scratch (reused across branches, sized by Dmax) ----
    half_t* xh   = alloc_h((size_t)B * Dmax * H);
    half_t* qh   = alloc_h((size_t)B * Dmax * H);  // (B,D,H) or (B,H,D)
    half_t* kh   = alloc_h((size_t)B * Dmax * H);
    half_t* vh   = alloc_h((size_t)B * Dmax * H);
    half_t* sc   = alloc_h((size_t)B * H * H);     // attention logits/probs
    half_t* tvsa = alloc_h((size_t)B * Dmax * H);  // (B,D,H)
    half_t* fvsa = alloc_h((size_t)B * Dmax * H);  // stored (B,H,D)
    half_t* twh  = alloc_h((size_t)B * Dmax * PW); // (B,D,PW), cols<100 valid
    half_t* fwh  = alloc_h((size_t)B * Dmax * PW);
    half_t* tfwh = alloc_h((size_t)B * 10000);     // (B,100,100) contiguous
    half_t* cat  = alloc_h((size_t)B * 5304);
    half_t* fc1h = alloc_h((size_t)B * 768);

    const float rsH = 1.0f / sqrtf((float)H);

    for (int bi = 0; bi < 3; ++bi) {
        const int D = Ds[bi];
        const float rsD = 1.0f / sqrtf((float)D);
        const float* tok = (const float*)d_in[bi];

        const long nx = (long)B * D * H;
        gather_x_k<<<(int)((nx + 255) / 256), 256, 0, stream>>>(tok, xh, D, H, nx);

        // ----- token-axis attention: flatten M = B*D over K = H -----
        gemm(stream, xh, H, 1, 0, Wtq[bi], H, 1, 0, btq[bi], qh, nullptr, H, 0, B * D, H, H, 1, 1.0f);
        gemm(stream, xh, H, 1, 0, Wtk[bi], H, 1, 0, btk[bi], kh, nullptr, H, 0, B * D, H, H, 1, 1.0f);
        gemm(stream, xh, H, 1, 0, Wtv[bi], H, 1, 0, btv[bi], vh, nullptr, H, 0, B * D, H, H, 1, 1.0f);
        // scores = q @ k^T (B operand k-contiguous)
        gemm(stream, qh, H, 1, (long)D * H, kh, 1, H, (long)D * H, nullptr,
             sc, nullptr, D, (long)D * D, D, D, H, B, rsH);
        softmax_rows_k<<<B * D, 256, 0, stream>>>(sc, D);
        gemm(stream, sc, D, 1, (long)D * D, vh, H, 1, (long)D * H, nullptr,
             tvsa, nullptr, H, (long)D * H, D, H, D, B, 1.0f);

        // ----- feature-axis attention on x^T (A m-contiguous transpose) -----
        gemm(stream, xh, 1, H, (long)D * H, Wfq[bi], D, 1, 0, bfq[bi], qh, nullptr, D, (long)H * D, H, D, D, B, 1.0f);
        gemm(stream, xh, 1, H, (long)D * H, Wfk[bi], D, 1, 0, bfk[bi], kh, nullptr, D, (long)H * D, H, D, D, B, 1.0f);
        gemm(stream, xh, 1, H, (long)D * H, Wfv[bi], D, 1, 0, bfv[bi], vh, nullptr, D, (long)H * D, H, D, D, B, 1.0f);
        gemm(stream, qh, D, 1, (long)H * D, kh, 1, D, (long)H * D, nullptr,
             sc, nullptr, H, (long)H * H, H, H, D, B, rsD);
        softmax_rows_k<<<B * H, 256, 0, stream>>>(sc, H);
        gemm(stream, sc, H, 1, (long)H * H, vh, D, 1, (long)H * D, nullptr,
             fvsa, nullptr, D, (long)H * D, H, D, H, B, 1.0f);   // fvsa stored (B,H,D)

        // ----- tw / fw projections (H -> 100, padded stride PW) -----
        gemm(stream, tvsa, H, 1, 0, WTW[bi], PW, 1, 0, bTW[bi], twh, nullptr, PW, 0, B * D, 100, H, 1, 1.0f);
        gemm(stream, fvsa, 1, D, (long)H * D, WFW[bi], PW, 1, 0, bFW[bi],
             fwh, nullptr, PW, (long)D * PW, D, 100, H, B, 1.0f);

        // ----- tfw[b,i,j] = sum_d tw[b,d,i] * fw[b,d,j] -----
        gemm(stream, twh, 1, PW, (long)D * PW, fwh, PW, 1, (long)D * PW, nullptr,
             tfwh, nullptr, 100, 10000, 100, 100, D, B, 1.0f);

        // ----- woven: (B x 10000) @ (10000 x 1000) -> concat columns -----
        gemm(stream, tfwh, 10000, 1, 0, Wwov[bi], 1000, 1, 0, bwov[bi],
             cat + 2304 + 1000 * bi, nullptr, 5304, 0, B, 1000, 10000, 1, 1.0f);
    }

    // cls + mask gathers into concat cols [0, 2304)
    const long ncat = (long)B * 2304;
    build_concat_k<<<(int)((ncat + 255) / 256), 256, 0, stream>>>(
        (const float*)d_in[0], (const float*)d_in[1], (const float*)d_in[2],
        (const int*)d_in[3], cat, ncat);

    // fc1: (B x 5304) @ (5304 x 768); fc2: (B x 768) @ (768 x 5) -> f32 out
    gemm(stream, cat, 5304, 1, 0, Wfc1, 768, 1, 0, bfc1, fc1h, nullptr, 768, 0, B, 768, 5304, 1, 1.0f);
    gemm(stream, fc1h, 768, 1, 0, Wfc2, 5, 1, 0, bfc2, nullptr, (float*)d_out, 5, 0, B, 5, 768, 1, 1.0f);
}
